// MPNN_10453950399097
// MI455X (gfx1250) — matrix operand order repaired and checked by reading, not compiled, not run on hardware
//
#include <hip/hip_runtime.h>
#include <hip/hip_bf16.h>

// ---------------- problem constants (from reference) ----------------
#define N_NODES 20000
#define N_EDGES 40000
#define HDIM    64
#define NGRAPH  1000
#define DIN     15
#define DE      5
#define EHID    128
#define HH      4096          // H*H
#define NOUT    12
#define STEPS   6

typedef unsigned short u16;
typedef __bf16 v16bf __attribute__((ext_vector_type(16)));
typedef float  v8f   __attribute__((ext_vector_type(8)));
typedef int    gvec4i __attribute__((vector_size(16)));   // matches builtin param type

union BFrag { v16bf v; uint4 q[2]; };

// CDNA5 async global->LDS copy path (ASYNCcnt), if toolchain exposes it
#if defined(__has_builtin)
#if __has_builtin(__builtin_amdgcn_global_load_async_to_lds_b128)
#define MPNN_ASYNC_LDS 1
#endif
#endif

// ---------------- small device helpers ----------------
__device__ __forceinline__ float bf2f(u16 u) {
  return __uint_as_float(((unsigned)u) << 16);
}
__device__ __forceinline__ u16 f2bf(float f) {           // RNE
  unsigned u = __float_as_uint(f);
  u += 0x7FFFu + ((u >> 16) & 1u);
  return (u16)(u >> 16);
}
__device__ __forceinline__ float sigmoidf(float x) { return 1.f / (1.f + expf(-x)); }
// order-preserving float<->uint encoding for atomic max
__device__ __forceinline__ unsigned fenc(float f) {
  unsigned u = __float_as_uint(f);
  return (u & 0x80000000u) ? ~u : (u | 0x80000000u);
}
__device__ __forceinline__ float fdec(unsigned u) {
  return __uint_as_float((u & 0x80000000u) ? (u & 0x7FFFFFFFu) : ~u);
}

// ---------------- trivial kernels ----------------
__global__ void mpnn_zero_f(float* p, int n) {
  int i = blockIdx.x * blockDim.x + threadIdx.x;
  if (i < n) p[i] = 0.f;
}

// h = relu(x @ lin0_w + lin0_b)   grid=N, block=64
__global__ void mpnn_lin0(const float* __restrict__ x, const float* __restrict__ w,
                          const float* __restrict__ b, float* __restrict__ h) {
  int n = blockIdx.x, t = threadIdx.x;
  __shared__ float xs[DIN];
  if (t < DIN) xs[t] = x[n * DIN + t];
  __syncthreads();
  float s = b[t];
#pragma unroll
  for (int i = 0; i < DIN; ++i) s = fmaf(xs[i], w[i * HDIM + t], s);
  h[n * HDIM + t] = fmaxf(s, 0.f);
}

// eh = relu(edge_attr @ en_w1 + en_b1) -> bf16   grid=E, block=128
__global__ void mpnn_edge_hidden(const float* __restrict__ ea, const float* __restrict__ w,
                                 const float* __restrict__ b, u16* __restrict__ eh) {
  int e = blockIdx.x, t = threadIdx.x;
  __shared__ float es[DE];
  if (t < DE) es[t] = ea[e * DE + t];
  __syncthreads();
  float s = b[t];
#pragma unroll
  for (int i = 0; i < DE; ++i) s = fmaf(es[i], w[i * EHID + t], s);
  eh[e * EHID + t] = f2bf(fmaxf(s, 0.f));
}

// en_w2[128,4096] -> en_w2t[4096,128] bf16
__global__ void mpnn_cvt_w2t(const float* __restrict__ w, u16* __restrict__ o) {
  int i = blockIdx.x * blockDim.x + threadIdx.x;
  if (i >= HH * EHID) return;
  int n = i / EHID, k = i % EHID;
  o[i] = f2bf(w[k * HH + n]);
}
// ng_w1[4096,128] -> ng_w1t[128,4096] bf16
__global__ void mpnn_cvt_w1t(const float* __restrict__ w, u16* __restrict__ o) {
  int i = blockIdx.x * blockDim.x + threadIdx.x;
  if (i >= EHID * HH) return;
  int n = i / HH, k = i % HH;
  o[i] = f2bf(w[k * EHID + n]);
}

// ---------------- WMMA GEMM 1: w_flat = eh @ en_w2 + en_b2  (bf16 out) ----------------
// A (10 MB) and B (1 MB) are L2-resident; kernel is bound by the 327 MB store.
// grid (E/16, 4096/64), block 128 (4 waves, one 16x16 N-tile each)
__global__ void __launch_bounds__(128)
mpnn_gemm_wflat(const u16* __restrict__ A,   // eh   [E,128] bf16 row-major
                const u16* __restrict__ Bt,  // w2t  [4096,128] bf16 (N-major)
                const float* __restrict__ bias,
                u16* __restrict__ C) {       // wflat [E,4096] bf16
  int lane = threadIdx.x & 31, wave = threadIdx.x >> 5;
  int row0 = blockIdx.x * 16;
  int col0 = blockIdx.y * 64 + wave * 16;
  int rsel = lane & 15, kgrp = lane >> 4;
  int rowA = row0 + rsel, ncol = col0 + rsel;

  v8f c;
  float bv = bias[ncol];
#pragma unroll
  for (int r = 0; r < 8; ++r) c[r] = bv;

  const u16* ap = A + (size_t)rowA * EHID + kgrp * 8;
  const u16* bp = Bt + (size_t)ncol * EHID + kgrp * 8;
#pragma unroll
  for (int kb = 0; kb < EHID; kb += 32) {
    BFrag a, b;
    a.q[0] = *(const uint4*)(ap + kb);
    a.q[1] = *(const uint4*)(ap + kb + 16);
    b.q[0] = *(const uint4*)(bp + kb);
    b.q[1] = *(const uint4*)(bp + kb + 16);
    c = __builtin_amdgcn_wmma_f32_16x16x32_bf16(false, a.v, false, b.v,
                                                (short)0, c, false, false);
  }
  int m0 = row0 + kgrp * 8;
#pragma unroll
  for (int r = 0; r < 8; ++r)
    C[(size_t)(m0 + r) * HH + ncol] = f2bf(c[r]);
}

// ---------------- WMMA GEMM 2: recon_h = relu(w_flat @ ng_w1 + ng_b1) ----------------
// A = w_flat (327 MB > L2): must be streamed from HBM exactly once.
// One 256-thread block (8 waves) owns a 16-row A tile and ALL 128 output
// columns; the A tile is staged through LDS in 512-wide K chunks so every
// HBM byte of w_flat is read once. B (1 MB) stays L2-resident.
#define KC      512
#define ASTRIDE (KC + 24)     // u16 units; row stride 1072 B: 16B-aligned, 12-bank stagger

__global__ void __launch_bounds__(256)
mpnn_gemm_reconh(const u16* __restrict__ A,   // wflat [E,4096] bf16
                 const u16* __restrict__ Bt,  // w1t   [128,4096] bf16 (N-major)
                 const float* __restrict__ bias,
                 float* __restrict__ C) {     // recon_h [E,128] f32
  __shared__ __align__(16) u16 As[16 * ASTRIDE];   // ~17 KB
  int tid = threadIdx.x;
  int lane = tid & 31, wave = tid >> 5;
  int row0 = blockIdx.x * 16;
  int rsel = lane & 15, kgrp = lane >> 4;
  int ncol = wave * 16 + rsel;

  v8f c;
  float bv = bias[ncol];
#pragma unroll
  for (int r = 0; r < 8; ++r) c[r] = bv;

  const u16* bp = Bt + (size_t)ncol * HH + kgrp * 8;

  for (int k0 = 0; k0 < HH; k0 += KC) {
    __syncthreads();                       // LDS buffer reuse
    // cooperative stage: 16 rows x KC bf16 = 1024 x 16B, 4 per thread
    for (int i = tid; i < 16 * KC / 8; i += 256) {
      int r = i >> 6;                      // / (KC/8)
      int kk = (i & 63) << 3;
      const u16* g = A + (size_t)(row0 + r) * HH + k0 + kk;
#if MPNN_ASYNC_LDS
      __builtin_amdgcn_global_load_async_to_lds_b128(
          (__attribute__((address_space(1))) gvec4i*)const_cast<u16*>(g),
          (__attribute__((address_space(3))) gvec4i*)&As[r * ASTRIDE + kk],
          0, 0);
#else
      *(uint4*)&As[r * ASTRIDE + kk] = *(const uint4*)g;
#endif
    }
#if MPNN_ASYNC_LDS
#if __has_builtin(__builtin_amdgcn_s_wait_asynccnt)
    __builtin_amdgcn_s_wait_asynccnt(0);
#else
    asm volatile("s_wait_asynccnt 0" ::: "memory");
#endif
#endif
    __syncthreads();
#pragma unroll 4
    for (int kb = 0; kb < KC; kb += 32) {
      BFrag a, b;
      a.q[0] = *(const uint4*)&As[rsel * ASTRIDE + kb + kgrp * 8];
      a.q[1] = *(const uint4*)&As[rsel * ASTRIDE + kb + kgrp * 8 + 16];
      b.q[0] = *(const uint4*)(bp + k0 + kb);
      b.q[1] = *(const uint4*)(bp + k0 + kb + 16);
      c = __builtin_amdgcn_wmma_f32_16x16x32_bf16(false, a.v, false, b.v,
                                                  (short)0, c, false, false);
    }
  }
  int m0 = row0 + kgrp * 8;
#pragma unroll
  for (int r = 0; r < 8; ++r)
    C[(size_t)(m0 + r) * EHID + ncol] = fmaxf(c[r], 0.f);
}

// recon = recon_h @ ng_w2 + ng_b2 ; loss += sum((recon-ea)^2)   block 256
__global__ void mpnn_recon_loss(const float* __restrict__ rh, const float* __restrict__ w2,
                                const float* __restrict__ b2, const float* __restrict__ ea,
                                float* __restrict__ loss) {
  int e = blockIdx.x * blockDim.x + threadIdx.x;
  float l = 0.f;
  if (e < N_EDGES) {
    float a[DE];
#pragma unroll
    for (int j = 0; j < DE; ++j) a[j] = b2[j];
    const float* r = rh + (size_t)e * EHID;
    for (int i = 0; i < EHID; ++i) {
      float v = r[i];
#pragma unroll
      for (int j = 0; j < DE; ++j) a[j] = fmaf(v, w2[i * DE + j], a[j]);
    }
#pragma unroll
    for (int j = 0; j < DE; ++j) {
      float d = a[j] - ea[e * DE + j];
      l += d * d;
    }
  }
  __shared__ float sh[256];
  sh[threadIdx.x] = l;
  __syncthreads();
  for (int s = 128; s > 0; s >>= 1) {
    if (threadIdx.x < s) sh[threadIdx.x] += sh[threadIdx.x + s];
    __syncthreads();
  }
  if (threadIdx.x == 0) atomicAdd(loss, sh[0]);
}

__global__ void mpnn_loss_out(const float* __restrict__ loss, float* __restrict__ out) {
  out[0] = loss[0] / (float)(N_EDGES * DE);
}

// degree
__global__ void mpnn_deg_count(const int* __restrict__ tgt, float* __restrict__ deg) {
  int e = blockIdx.x * blockDim.x + threadIdx.x;
  if (e < N_EDGES) atomicAdd(&deg[tgt[e]], 1.f);
}
__global__ void mpnn_invdeg(const float* __restrict__ deg, float* __restrict__ inv) {
  int n = blockIdx.x * blockDim.x + threadIdx.x;
  if (n < N_NODES) inv[n] = 1.f / fmaxf(deg[n], 1.f);
}

// msg[e] = h[src[e]] @ We[e]; scatter-add to acc[tgt[e]]   grid=E, block=64
__global__ void mpnn_msg(const float* __restrict__ h, const u16* __restrict__ We,
                         const int* __restrict__ src, const int* __restrict__ tgt,
                         float* __restrict__ acc) {
  int e = blockIdx.x, t = threadIdx.x;
  __shared__ float sh[HDIM];
  sh[t] = h[src[e] * HDIM + t];
  __syncthreads();
  const u16* W = We + (size_t)e * HH + t;
  float m = 0.f;
#pragma unroll 8
  for (int i = 0; i < HDIM; ++i) m = fmaf(sh[i], bf2f(W[i * HDIM]), m);
  atomicAdd(&acc[tgt[e] * HDIM + t], m);
}

// GRU step   grid=N, block=64
__global__ void mpnn_gru(const float* __restrict__ acc, const float* __restrict__ invdeg,
                         const float* __restrict__ conv_b,
                         const float* __restrict__ wih, const float* __restrict__ whh,
                         const float* __restrict__ bih, const float* __restrict__ bhh,
                         float* __restrict__ h) {
  int n = blockIdx.x, t = threadIdx.x;
  __shared__ float sm[HDIM], shh[HDIM];
  float hprev = h[n * HDIM + t];
  sm[t] = fmaxf(acc[n * HDIM + t] * invdeg[n] + conv_b[t], 0.f);
  shh[t] = hprev;
  __syncthreads();
  float ir = bih[t], iz = bih[HDIM + t], in_ = bih[2 * HDIM + t];
  float hr = bhh[t], hz = bhh[HDIM + t], hn = bhh[2 * HDIM + t];
  for (int i = 0; i < HDIM; ++i) {
    float m = sm[i], hv = shh[i];
    ir = fmaf(m, wih[i * 192 + t], ir);
    iz = fmaf(m, wih[i * 192 + HDIM + t], iz);
    in_ = fmaf(m, wih[i * 192 + 2 * HDIM + t], in_);
    hr = fmaf(hv, whh[i * 192 + t], hr);
    hz = fmaf(hv, whh[i * 192 + HDIM + t], hz);
    hn = fmaf(hv, whh[i * 192 + 2 * HDIM + t], hn);
  }
  float r = sigmoidf(ir + hr);
  float z = sigmoidf(iz + hz);
  float nn = tanhf(in_ + r * hn);
  h[n * HDIM + t] = (1.f - z) * nn + z * hprev;
}

// Set2Set LSTM step   grid=B, block=64   (torch gate order i,f,g,o)
__global__ void mpnn_lstm(const float* __restrict__ q, const float* __restrict__ wih,
                          const float* __restrict__ whh, const float* __restrict__ bih,
                          const float* __restrict__ bhh,
                          float* __restrict__ hs, float* __restrict__ cs) {
  int b = blockIdx.x, t = threadIdx.x;
  __shared__ float qs[2 * HDIM], hl[HDIM];
  qs[t] = q[b * 2 * HDIM + t];
  qs[HDIM + t] = q[b * 2 * HDIM + HDIM + t];
  hl[t] = hs[b * HDIM + t];
  __syncthreads();
  float gi = bih[t] + bhh[t];
  float gf = bih[HDIM + t] + bhh[HDIM + t];
  float gg = bih[2 * HDIM + t] + bhh[2 * HDIM + t];
  float go = bih[3 * HDIM + t] + bhh[3 * HDIM + t];
  for (int i = 0; i < 2 * HDIM; ++i) {
    float v = qs[i];
    gi = fmaf(v, wih[i * 256 + t], gi);
    gf = fmaf(v, wih[i * 256 + HDIM + t], gf);
    gg = fmaf(v, wih[i * 256 + 2 * HDIM + t], gg);
    go = fmaf(v, wih[i * 256 + 3 * HDIM + t], go);
  }
  for (int i = 0; i < HDIM; ++i) {
    float v = hl[i];
    gi = fmaf(v, whh[i * 256 + t], gi);
    gf = fmaf(v, whh[i * 256 + HDIM + t], gf);
    gg = fmaf(v, whh[i * 256 + 2 * HDIM + t], gg);
    go = fmaf(v, whh[i * 256 + 3 * HDIM + t], go);
  }
  float c = sigmoidf(gf) * cs[b * HDIM + t] + sigmoidf(gi) * tanhf(gg);
  hs[b * HDIM + t] = sigmoidf(go) * tanhf(c);
  cs[b * HDIM + t] = c;
}

// e[n] = dot(h[n], hs[batch[n]]); segment max  grid=N, block=64
__global__ void mpnn_attn_e(const float* __restrict__ h, const float* __restrict__ hs,
                            const int* __restrict__ batch,
                            float* __restrict__ e_node, unsigned* __restrict__ emax) {
  int n = blockIdx.x, t = threadIdx.x;
  int b = batch[n];
  __shared__ float sh[HDIM];
  sh[t] = h[n * HDIM + t] * hs[b * HDIM + t];
  __syncthreads();
  for (int s = 32; s > 0; s >>= 1) {
    if (t < s) sh[t] += sh[t + s];
    __syncthreads();
  }
  if (t == 0) {
    e_node[n] = sh[0];
    atomicMax(&emax[b], fenc(sh[0]));
  }
}

// a = exp(e - emax[batch]); segment sum
__global__ void mpnn_attn_a(const float* __restrict__ e_node, const unsigned* __restrict__ emax,
                            const int* __restrict__ batch,
                            float* __restrict__ a_node, float* __restrict__ asum) {
  int n = blockIdx.x * blockDim.x + threadIdx.x;
  if (n >= N_NODES) return;
  int b = batch[n];
  float a = expf(e_node[n] - fdec(emax[b]));
  a_node[n] = a;
  atomicAdd(&asum[b], a);
}

// rr[b] += (a/asum) * h[n]   grid=N, block=64
__global__ void mpnn_attn_rr(const float* __restrict__ h, const float* __restrict__ a_node,
                             const float* __restrict__ asum, const int* __restrict__ batch,
                             float* __restrict__ rr) {
  int n = blockIdx.x, t = threadIdx.x;
  int b = batch[n];
  float w = a_node[n] / asum[b];
  atomicAdd(&rr[b * HDIM + t], w * h[n * HDIM + t]);
}

// q_star = [hs, rr]   grid=B, block=128
__global__ void mpnn_qstar(const float* __restrict__ hs, const float* __restrict__ rr,
                           float* __restrict__ q) {
  int b = blockIdx.x, t = threadIdx.x;
  q[b * 2 * HDIM + t] = (t < HDIM) ? hs[b * HDIM + t] : rr[b * HDIM + (t - HDIM)];
}

// y = relu(q_star @ lin1 + b1) @ lin2 + b2   grid=B, block=64
__global__ void mpnn_final(const float* __restrict__ q, const float* __restrict__ w1,
                           const float* __restrict__ b1, const float* __restrict__ w2,
                           const float* __restrict__ b2, float* __restrict__ y) {
  int b = blockIdx.x, t = threadIdx.x;
  __shared__ float qs[2 * HDIM], u[HDIM];
  qs[t] = q[b * 2 * HDIM + t];
  qs[HDIM + t] = q[b * 2 * HDIM + HDIM + t];
  __syncthreads();
  float s = b1[t];
  for (int i = 0; i < 2 * HDIM; ++i) s = fmaf(qs[i], w1[i * HDIM + t], s);
  u[t] = fmaxf(s, 0.f);
  __syncthreads();
  if (t < NOUT) {
    float o = b2[t];
    for (int i = 0; i < HDIM; ++i) o = fmaf(u[i], w2[i * NOUT + t], o);
    y[b * NOUT + t] = o;
  }
}

// ---------------- host side ----------------
static inline char* wsa(char*& p, size_t bytes) {
  char* r = p;
  p += (bytes + 255) & ~(size_t)255;
  return r;
}

extern "C" void kernel_launch(void* const* d_in, const int* in_sizes, int n_in,
                              void* d_out, int out_size, void* d_ws, size_t ws_size,
                              hipStream_t stream) {
  const float* x        = (const float*)d_in[0];
  const int*   ei       = (const int*)d_in[1];
  const float* ea       = (const float*)d_in[2];
  const int*   batch    = (const int*)d_in[3];
  const float* lin0_w   = (const float*)d_in[4];
  const float* lin0_b   = (const float*)d_in[5];
  const float* en_w1    = (const float*)d_in[6];
  const float* en_b1    = (const float*)d_in[7];
  const float* en_w2    = (const float*)d_in[8];
  const float* en_b2    = (const float*)d_in[9];
  const float* ng_w1    = (const float*)d_in[10];
  const float* ng_b1    = (const float*)d_in[11];
  const float* ng_w2    = (const float*)d_in[12];
  const float* ng_b2    = (const float*)d_in[13];
  const float* conv_b   = (const float*)d_in[14];
  const float* gru_wih  = (const float*)d_in[15];
  const float* gru_whh  = (const float*)d_in[16];
  const float* gru_bih  = (const float*)d_in[17];
  const float* gru_bhh  = (const float*)d_in[18];
  const float* s2s_wih  = (const float*)d_in[19];
  const float* s2s_whh  = (const float*)d_in[20];
  const float* s2s_bih  = (const float*)d_in[21];
  const float* s2s_bhh  = (const float*)d_in[22];
  const float* lin1_w   = (const float*)d_in[23];
  const float* lin1_b   = (const float*)d_in[24];
  const float* lin2_w   = (const float*)d_in[25];
  const float* lin2_b   = (const float*)d_in[26];
  float* out = (float*)d_out;

  const int* src = ei;
  const int* tgt = ei + N_EDGES;

  char* p = (char*)d_ws;
  float* h      = (float*)wsa(p, (size_t)N_NODES * HDIM * 4);
  u16*   eh     = (u16*)  wsa(p, (size_t)N_EDGES * EHID * 2);
  u16*   w2t    = (u16*)  wsa(p, (size_t)HH * EHID * 2);
  u16*   w1t    = (u16*)  wsa(p, (size_t)EHID * HH * 2);
  u16*   wflat  = (u16*)  wsa(p, (size_t)N_EDGES * HH * 2);
  float* reconh = (float*)wsa(p, (size_t)N_EDGES * EHID * 4);
  float* acc    = (float*)wsa(p, (size_t)N_NODES * HDIM * 4);
  float* deg    = (float*)wsa(p, (size_t)N_NODES * 4);
  float* invdeg = (float*)wsa(p, (size_t)N_NODES * 4);
  float* loss   = (float*)wsa(p, 256);
  float* hs     = (float*)wsa(p, (size_t)NGRAPH * HDIM * 4);
  float* cs     = (float*)wsa(p, (size_t)NGRAPH * HDIM * 4);
  float* qstar  = (float*)wsa(p, (size_t)NGRAPH * 2 * HDIM * 4);
  float* e_node = (float*)wsa(p, (size_t)N_NODES * 4);
  float* a_node = (float*)wsa(p, (size_t)N_NODES * 4);
  unsigned* emax = (unsigned*)wsa(p, (size_t)NGRAPH * 4);
  float* asum   = (float*)wsa(p, (size_t)NGRAPH * 4);
  float* rr     = (float*)wsa(p, (size_t)NGRAPH * HDIM * 4);

  // ---- node embedding + edge network ----
  mpnn_zero_f<<<1, 64, 0, stream>>>(loss, 1);
  mpnn_lin0<<<N_NODES, HDIM, 0, stream>>>(x, lin0_w, lin0_b, h);
  mpnn_edge_hidden<<<N_EDGES, EHID, 0, stream>>>(ea, en_w1, en_b1, eh);
  mpnn_cvt_w2t<<<(HH * EHID + 255) / 256, 256, 0, stream>>>(en_w2, w2t);
  mpnn_cvt_w1t<<<(EHID * HH + 255) / 256, 256, 0, stream>>>(ng_w1, w1t);

  mpnn_gemm_wflat<<<dim3(N_EDGES / 16, HH / 64), 128, 0, stream>>>(eh, w2t, en_b2, wflat);
  mpnn_gemm_reconh<<<N_EDGES / 16, 256, 0, stream>>>(wflat, w1t, ng_b1, reconh);
  mpnn_recon_loss<<<(N_EDGES + 255) / 256, 256, 0, stream>>>(reconh, ng_w2, ng_b2, ea, loss);
  mpnn_loss_out<<<1, 1, 0, stream>>>(loss, out + (size_t)NGRAPH * NOUT);

  // ---- degree ----
  mpnn_zero_f<<<(N_NODES + 255) / 256, 256, 0, stream>>>(deg, N_NODES);
  mpnn_deg_count<<<(N_EDGES + 255) / 256, 256, 0, stream>>>(tgt, deg);
  mpnn_invdeg<<<(N_NODES + 255) / 256, 256, 0, stream>>>(deg, invdeg);

  // ---- message passing + GRU ----
  for (int s = 0; s < STEPS; ++s) {
    mpnn_zero_f<<<(N_NODES * HDIM + 255) / 256, 256, 0, stream>>>(acc, N_NODES * HDIM);
    mpnn_msg<<<N_EDGES, HDIM, 0, stream>>>(h, wflat, src, tgt, acc);
    mpnn_gru<<<N_NODES, HDIM, 0, stream>>>(acc, invdeg, conv_b, gru_wih, gru_whh,
                                           gru_bih, gru_bhh, h);
  }

  // ---- Set2Set ----
  mpnn_zero_f<<<(NGRAPH * HDIM + 255) / 256, 256, 0, stream>>>(hs, NGRAPH * HDIM);
  mpnn_zero_f<<<(NGRAPH * HDIM + 255) / 256, 256, 0, stream>>>(cs, NGRAPH * HDIM);
  mpnn_zero_f<<<(NGRAPH * 2 * HDIM + 255) / 256, 256, 0, stream>>>(qstar, NGRAPH * 2 * HDIM);
  for (int s = 0; s < STEPS; ++s) {
    mpnn_zero_f<<<(NGRAPH + 255) / 256, 256, 0, stream>>>((float*)emax, NGRAPH); // 0 == enc(<-inf)
    mpnn_zero_f<<<(NGRAPH + 255) / 256, 256, 0, stream>>>(asum, NGRAPH);
    mpnn_zero_f<<<(NGRAPH * HDIM + 255) / 256, 256, 0, stream>>>(rr, NGRAPH * HDIM);
    mpnn_lstm<<<NGRAPH, HDIM, 0, stream>>>(qstar, s2s_wih, s2s_whh, s2s_bih, s2s_bhh, hs, cs);
    mpnn_attn_e<<<N_NODES, HDIM, 0, stream>>>(h, hs, batch, e_node, emax);
    mpnn_attn_a<<<(N_NODES + 255) / 256, 256, 0, stream>>>(e_node, emax, batch, a_node, asum);
    mpnn_attn_rr<<<N_NODES, HDIM, 0, stream>>>(h, a_node, asum, batch, rr);
    mpnn_qstar<<<NGRAPH, 2 * HDIM, 0, stream>>>(hs, rr, qstar);
  }

  // ---- readout ----
  mpnn_final<<<NGRAPH, HDIM, 0, stream>>>(qstar, lin1_w, lin1_b, lin2_w, lin2_b, out);
}
